// AdditiveAttention_57543971832359
// MI455X (gfx1250) — matrix operand order repaired and checked by reading, not compiled, not run on hardware
//
#include <hip/hip_runtime.h>
#include <hip/hip_bf16.h>

#define BS  4
#define NQ  1024
#define NK  1024
#define QD  128
#define KD  128
#define VD  128
#define HID 64

typedef __attribute__((ext_vector_type(16))) __bf16 v16bf;
typedef __attribute__((ext_vector_type(8)))  float  v8f;

union ABfrag { uint4 u[2]; v16bf v; };

// Prefer the CDNA5 hardware transcendental v_tanh_f32 if the toolchain
// exposes it; fall back to a 1-exp + 1-div formulation otherwise.
__device__ __forceinline__ float fast_tanh(float x) {
#if __has_builtin(__builtin_amdgcn_tanhf)
    return __builtin_amdgcn_tanhf(x);
#else
    float e = __expf(-2.0f * fabsf(x));
    float t = (1.0f - e) / (1.0f + e);
    return copysignf(t, x);
#endif
}

// ---------------------------------------------------------------------------
// K0: row-wise projection.  blockIdx < BS*NQ -> queries @ W_q, else keys @ W_k.
// 64 threads; input row (128 f32) staged in LDS; thread h computes output h.
// ---------------------------------------------------------------------------
__global__ void proj_kernel(const float* __restrict__ queries,
                            const float* __restrict__ keys,
                            const float* __restrict__ W_q,
                            const float* __restrict__ W_k,
                            float* __restrict__ q_proj,
                            float* __restrict__ k_proj) {
    __shared__ float sh[QD];
    int r = blockIdx.x;
    const float* src; const float* W; float* dst;
    if (r < BS * NQ) {
        src = queries + (size_t)r * QD;  W = W_q;  dst = q_proj + (size_t)r * HID;
    } else {
        int rr = r - BS * NQ;
        src = keys + (size_t)rr * KD;    W = W_k;  dst = k_proj + (size_t)rr * HID;
    }
    int tid = threadIdx.x;                 // 0..63
    sh[tid]      = src[tid];
    sh[tid + 64] = src[tid + 64];
    __syncthreads();
    float acc = 0.0f;
#pragma unroll 8
    for (int d = 0; d < QD; ++d)
        acc += sh[d] * W[d * HID + tid];   // coalesced across the 64 threads
    dst[tid] = acc;
}

// ---------------------------------------------------------------------------
// K1: values (f32, [b][k][n]) -> vT (bf16, [b][n][k]).  Transposed so the WMMA
// B-operand (a column of V) is a contiguous run of K in memory.
// ---------------------------------------------------------------------------
__global__ void vconv_kernel(const float* __restrict__ values,
                             __hip_bfloat16* __restrict__ vT) {
    int idx = blockIdx.x * blockDim.x + threadIdx.x;   // 0 .. BS*NK*VD-1
    int n = idx % VD;
    int k = (idx / VD) % NK;
    int b = idx / (VD * NK);
    vT[((size_t)b * VD + n) * NK + k] = __float2bfloat16(values[idx]);
}

// ---------------------------------------------------------------------------
// K2: fused scores + softmax for one (b, i) query row per 256-thread block.
// Each thread owns 4 key columns; 64 tanh-FMAs each; block reduce max & sum;
// write normalized probabilities as bf16 into P.
// Outer key loop kept rolled (unroll 1) so only one row of 16 float4 loads is
// in flight at a time -> no scratch spill.
// ---------------------------------------------------------------------------
__global__ void score_softmax_kernel(const float* __restrict__ q_proj,
                                     const float* __restrict__ k_proj,
                                     const float* __restrict__ w_v,
                                     __hip_bfloat16* __restrict__ P) {
    __shared__ float shq[HID];
    __shared__ float shw[HID];
    __shared__ float red[256];
    int b = blockIdx.x / NQ;
    int i = blockIdx.x % NQ;
    int tid = threadIdx.x;

    if (tid < HID) {
        shq[tid] = q_proj[((size_t)b * NQ + i) * HID + tid];
        shw[tid] = w_v[tid];
    }
    __syncthreads();

    float s[4];
    float lmax = -1e30f;
#pragma unroll 1
    for (int jj = 0; jj < 4; ++jj) {
        int j = tid + jj * 256;
        const float4* kp = (const float4*)(k_proj + ((size_t)b * NK + j) * HID);
        float acc = 0.0f;
#pragma unroll
        for (int h4 = 0; h4 < HID / 4; ++h4) {
            float4 kv = kp[h4];
            int h = h4 * 4;
            acc += shw[h + 0] * fast_tanh(shq[h + 0] + kv.x);
            acc += shw[h + 1] * fast_tanh(shq[h + 1] + kv.y);
            acc += shw[h + 2] * fast_tanh(shq[h + 2] + kv.z);
            acc += shw[h + 3] * fast_tanh(shq[h + 3] + kv.w);
        }
        s[jj] = acc;
        lmax  = fmaxf(lmax, acc);
    }

    // block max
    red[tid] = lmax; __syncthreads();
    for (int off = 128; off > 0; off >>= 1) {
        if (tid < off) red[tid] = fmaxf(red[tid], red[tid + off]);
        __syncthreads();
    }
    float rowmax = red[0];
    __syncthreads();

    float e[4];
    float lsum = 0.0f;
#pragma unroll
    for (int jj = 0; jj < 4; ++jj) { e[jj] = __expf(s[jj] - rowmax); lsum += e[jj]; }

    // block sum
    red[tid] = lsum; __syncthreads();
    for (int off = 128; off > 0; off >>= 1) {
        if (tid < off) red[tid] += red[tid + off];
        __syncthreads();
    }
    float inv = 1.0f / red[0];

    __hip_bfloat16* prow = P + ((size_t)b * NQ + i) * NK;
#pragma unroll
    for (int jj = 0; jj < 4; ++jj)
        prow[tid + jj * 256] = __float2bfloat16(e[jj] * inv);
}

// ---------------------------------------------------------------------------
// K3: out = P @ V via v_wmma_f32_16x16x32_bf16.
// grid = BS * NQ/16 blocks of 256 threads (8 waves); wave w owns output tile
// (qt*16 .. qt*16+15) x (w*16 .. w*16+15); K swept 32 at a time.
//
// A (16x32 bf16) per-lane layout (ISA 7.12.2): lane<16 -> row=lane, K 0-7 in
// v0-3, K 16-23 in v4-7; lane>=16 -> row=lane-16, K 8-15 / 24-31.
// B (32x16 bf16): lanes 0-15 -> col=lane, K 0-7 in v0-3, K 16-23 in v4-7; lanes 16-31 ->
// col=lane-16, K 8-15 / 24-31.  vT is [n][k] so each lane's 8-K runs are contiguous.
// ---------------------------------------------------------------------------
__global__ void pv_wmma_kernel(const __hip_bfloat16* __restrict__ P,
                               const __hip_bfloat16* __restrict__ vT,
                               float* __restrict__ out) {
    int b    = blockIdx.x >> 6;       // / (NQ/16)
    int qt   = blockIdx.x & 63;
    int w    = threadIdx.x >> 5;      // v-tile 0..7
    int lane = threadIdx.x & 31;
    int row  = lane & 15;
    int hi   = lane >> 4;

    const __hip_bfloat16* Pr = P  + ((size_t)b * NQ + qt * 16 + row) * NK; // A row
    const __hip_bfloat16* Vr = vT + ((size_t)b * VD + w  * 16 + row) * NK; // B col (row of vT)

    v8f acc = {};
#pragma unroll 4
    for (int kt = 0; kt < NK; kt += 32) {
        // speculative prefetch of the next K-slab (global_prefetch_b8)
        __builtin_prefetch(Pr + kt + 32, 0, 1);
        __builtin_prefetch(Vr + kt + 32, 0, 1);
        ABfrag a, bb;
        a.u[0]  = *(const uint4*)(Pr + kt + 8 * hi);        // K 0-7  / 8-15
        a.u[1]  = *(const uint4*)(Pr + kt + 8 * hi + 16);   // K 16-23 / 24-31
        bb.u[0] = *(const uint4*)(Vr + kt + 8 * hi);        // K 0-7  / 8-15
        bb.u[1] = *(const uint4*)(Vr + kt + 8 * hi + 16);   // K 16-23 / 24-31
        acc = __builtin_amdgcn_wmma_f32_16x16x32_bf16(
            /*neg_a=*/false, a.v, /*neg_b=*/false, bb.v,
            /*c_mod=*/(short)0, acc, /*reuse_a=*/false, /*reuse_b=*/false);
    }

    // D layout: VGPR r -> M = r + 8*hi, N = lane&15
    float* obase = out + ((size_t)b * NQ + qt * 16) * VD + w * 16 + row;
#pragma unroll
    for (int r = 0; r < 8; ++r)
        obase[(size_t)(r + 8 * hi) * VD] = acc[r];
}

// ---------------------------------------------------------------------------
extern "C" void kernel_launch(void* const* d_in, const int* in_sizes, int n_in,
                              void* d_out, int out_size, void* d_ws, size_t ws_size,
                              hipStream_t stream) {
    const float* queries = (const float*)d_in[0];
    const float* keys    = (const float*)d_in[1];
    const float* values  = (const float*)d_in[2];
    const float* W_q     = (const float*)d_in[3];
    const float* W_k     = (const float*)d_in[4];
    const float* w_v     = (const float*)d_in[5];
    float* out = (float*)d_out;

    // Workspace layout (all 1 MB aligned; total 11 MB, L2-resident):
    //   [0,1MB)   q_proj  f32  [BS*NQ][HID]
    //   [1,2MB)   k_proj  f32  [BS*NK][HID]
    //   [2,10MB)  P       bf16 [BS*NQ][NK]
    //   [10,11MB) vT      bf16 [BS*VD][NK]
    char* ws = (char*)d_ws;
    float*          q_proj = (float*)(ws);
    float*          k_proj = (float*)(ws + (1u  << 20));
    __hip_bfloat16* P      = (__hip_bfloat16*)(ws + (2u  << 20));
    __hip_bfloat16* vT     = (__hip_bfloat16*)(ws + (10u << 20));

    proj_kernel<<<2 * BS * NQ, 64, 0, stream>>>(queries, keys, W_q, W_k, q_proj, k_proj);
    vconv_kernel<<<(BS * NK * VD) / 256, 256, 0, stream>>>(values, vT);
    score_softmax_kernel<<<BS * NQ, 256, 0, stream>>>(q_proj, k_proj, w_v, P);
    pv_wmma_kernel<<<BS * (NQ / 16), 256, 0, stream>>>(P, vT, out);
}
